// ReformerWithCustomEmbeddings_19842748907661
// MI455X (gfx1250) — matrix operand order repaired and checked by reading, not compiled, not run on hardware
//
#include <hip/hip_runtime.h>
#include <hip/hip_bf16.h>

// ---------------------------------------------------------------------------
// Reformer forward on gfx1250 (MI455X).
// Heavy GEMMs: v_wmma_f32_16x16x32_bf16, double-buffered LDS pipeline.
//   - A tile staged with CDNA5 async DMA: global_load_async_to_lds_b128
//     (ASYNCcnt, waited with s_wait_asynccnt before the consuming barrier).
//   - B tile transposed into LDS (Bs[n][k]) with packed b32 stores (static
//     element indices: dynamic vector extracts would force a VGPR spill) so
//     both fragments read as 2x ds_load_b128 per lane.
//   - 4 accumulators per wave (16x64) amortize the A fragment; next tile is
//     prefetched with global_prefetch_b8.
//   - Epilogue fully templated (no runtime branches).
// Attention (64x128x64 per chunk) is VALU; hashing + per-(b,h) bitonic sort
// in LDS. Output scatter at original positions implements the "unsort" free.
// ---------------------------------------------------------------------------

typedef __bf16 bf16_t;
typedef __attribute__((ext_vector_type(16))) __bf16 v16bf;
typedef __attribute__((ext_vector_type(8)))  float  v8f;

// Model constants (from the reference)
constexpr int kV  = 16384;
constexpr int kD  = 768;
constexpr int kH  = 12;
constexpr int kDH = 64;
constexpr int kL  = 2;
constexpr int kF  = 3072;
constexpr int kC  = 64;
constexpr int kNBh = 32;         // NB/2 rotation dims
constexpr int kB  = 2;
constexpr int kS  = 2048;
constexpr int kNC = kS / kC;     // 32 chunks
constexpr int kM  = kB * kS;     // 4096 rows

// GEMM LDS geometry (bytes)
constexpr int kAB = 64 * 40 * 2;    // As buffer: 64 rows x (32+8 pad) bf16
constexpr int kBB = 128 * 40 * 2;   // Bs buffer: 128 cols x (32+8 pad) bf16

struct __attribute__((aligned(16))) B8 { bf16_t v[8]; };

// ---------------------------------------------------------------------------
// f32 -> bf16 conversion (grid-stride)
// ---------------------------------------------------------------------------
__global__ void k_f32_to_bf16(const float* __restrict__ in, bf16_t* __restrict__ out, long n) {
  long i = (long)blockIdx.x * blockDim.x + threadIdx.x;
  long stride = (long)gridDim.x * blockDim.x;
  for (; i < n; i += stride) out[i] = (bf16_t)in[i];
}

// ---------------------------------------------------------------------------
// Embedding gather + positional add: x[b,s,:] = emb[id] + pos[s]
// ---------------------------------------------------------------------------
__global__ void k_embed(const int* __restrict__ ids, const float* __restrict__ emb,
                        const float* __restrict__ pos, float* __restrict__ x) {
  int row = blockIdx.x;                  // 0..M-1
  int s = row % kS;
  int id = ids[row];
  const float* e = emb + (long)id * kD;
  const float* p = pos + (long)s * kD;
  float* o = x + (long)row * kD;
  for (int d = threadIdx.x; d < kD; d += blockDim.x) o[d] = e[d] + p[d];
}

// ---------------------------------------------------------------------------
// LayerNorm (one 256-thread block per row, eps = 1e-12, mean/var over D=768)
// ---------------------------------------------------------------------------
__global__ void k_layernorm(const float* __restrict__ x, const float* __restrict__ g,
                            const float* __restrict__ bb, float* __restrict__ y32,
                            bf16_t* __restrict__ y16) {
  int row = blockIdx.x;
  const float* xr = x + (long)row * kD;
  float s1 = 0.f, s2 = 0.f;
  for (int d = threadIdx.x; d < kD; d += blockDim.x) { float v = xr[d]; s1 += v; s2 += v * v; }
  __shared__ float r1[256], r2[256];
  r1[threadIdx.x] = s1; r2[threadIdx.x] = s2;
  __syncthreads();
  for (int o = 128; o > 0; o >>= 1) {
    if ((int)threadIdx.x < o) { r1[threadIdx.x] += r1[threadIdx.x + o]; r2[threadIdx.x] += r2[threadIdx.x + o]; }
    __syncthreads();
  }
  float mean = r1[0] / (float)kD;
  float var  = r2[0] / (float)kD - mean * mean;
  float inv  = rsqrtf(var + 1e-12f);
  for (int d = threadIdx.x; d < kD; d += blockDim.x) {
    float v = (xr[d] - mean) * inv * g[d] + bb[d];
    if (y32) y32[(long)row * kD + d] = v;
    if (y16) y16[(long)row * kD + d] = (bf16_t)v;
  }
}

// ---------------------------------------------------------------------------
// bf16 WMMA GEMM: out[M,N] = A[M,K] * W[K,N] (+bias)(+gelu)(+resid).
// Block = 256 threads = 8 waves; tile 64x128, 4x2 wave grid, each wave 16x64
// (4 accumulators). K-step 32. Double-buffered LDS software pipeline:
//   iter t: s_wait_asynccnt; barrier; stage(t+1) -> other buffer; compute(t).
// A staged via global_load_async_to_lds_b128 (unified smem block => LDS
// offsets are plain integers, base 0 for the sole allocation).
// B staged via 2x coalesced 16B reads + 8 packed ds_store_b32 (transpose).
// ---------------------------------------------------------------------------
template <bool BIAS, bool GELU, bool RESID, bool O32, bool O16>
__launch_bounds__(256)
__global__ void k_gemm_bf16(const bf16_t* __restrict__ A, const bf16_t* __restrict__ W,
                            const float* __restrict__ bias, const float* __restrict__ resid,
                            float* __restrict__ out32, bf16_t* __restrict__ out16,
                            int M, int N, int K) {
  __shared__ char smem[2 * kAB + 2 * kBB];   // [As0|As1|Bs0|Bs1], 30 KB
  int tid  = threadIdx.x;
  int wave = tid >> 5, lane = tid & 31;
  int wm = wave >> 1, wn = wave & 1;         // 4 x 2 wave grid
  int half = lane >> 4, l16 = lane & 15;
  long bm = (long)blockIdx.x * 64;
  long bn = (long)blockIdx.y * 128;

  // A staging: one 16B chunk per thread (row ar, cols ac..ac+7)
  int ar = tid >> 2;
  int ac = (tid & 3) * 8;
  // B staging: two k-rows (k2,k2+1) x 8 n-cols per thread
  int n8 = (tid & 15) * 8;
  int k2 = (tid >> 4) * 2;

  const bf16_t* aptr  = A + (bm + ar) * (long)K + ac;
  const bf16_t* wptr0 = W + (long)k2 * N + bn + n8;
  const bf16_t* wptr1 = wptr0 + N;
  unsigned aoff = (unsigned)(ar * 80 + ac * 2);    // byte offset inside an As buffer

  auto stage = [&](int k0, int buf) {
    // ---- A: async DMA global -> LDS (16B per lane), tracked by ASYNCcnt
    unsigned dst = (unsigned)(buf * kAB) + aoff;
    unsigned long long ga = (unsigned long long)(aptr + k0);
    asm volatile("global_load_async_to_lds_b128 %0, %1, off" :: "v"(dst), "v"(ga) : "memory");
    // ---- B: coalesced reads, packed-pair transpose scatter (8x ds_store_b32).
    // NOTE: static element index i (a lane-dependent index would be a dynamic
    // VGPR extract -> memory spill). ~8-way store bank conflict accepted; the
    // stores are overlapped with WMMA by the pipeline.
    char* bsb = smem + 2 * kAB + buf * kBB;
    B8 t0 = *(const B8*)(wptr0 + (long)k0 * N);
    B8 t1 = *(const B8*)(wptr1 + (long)k0 * N);
#pragma unroll
    for (int i = 0; i < 8; i++) {
      union { bf16_t h[2]; unsigned u; } p;
      p.h[0] = t0.v[i]; p.h[1] = t1.v[i];
      *(unsigned*)(bsb + 80 * (n8 + i) + 2 * k2) = p.u;
    }
    // ---- prefetch the tile after next (global_prefetch_b8)
    if (k0 + 32 < K) {
      __builtin_prefetch(aptr + k0 + 32, 0, 1);
      __builtin_prefetch(wptr0 + (long)(k0 + 32) * N, 0, 1);
    }
  };

  v8f acc[4];
#pragma unroll
  for (int t = 0; t < 4; t++) acc[t] = (v8f){0.f, 0.f, 0.f, 0.f, 0.f, 0.f, 0.f, 0.f};

  int nt = K / 32;
  stage(0, 0);
  for (int t = 0; t < nt; t++) {
    int cur = t & 1;
    asm volatile("s_wait_asynccnt 0" ::: "memory");  // our A DMA for tile t landed
    __syncthreads();                                 // all waves' staging visible
    if (t + 1 < nt) stage((t + 1) * 32, 1 - cur);    // fill other buffer (free now)

    const bf16_t* As = (const bf16_t*)(smem + cur * kAB);
    const bf16_t* Bs = (const bf16_t*)(smem + 2 * kAB + cur * kBB);

    v16bf a;
    const bf16_t* arow = As + (wm * 16 + l16) * 40;
#pragma unroll
    for (int j = 0; j < 8; j++) a[j] = arow[half * 8 + j];
#pragma unroll
    for (int j = 0; j < 8; j++) a[j + 8] = arow[16 + half * 8 + j];

#pragma unroll
    for (int tt = 0; tt < 4; tt++) {
      const bf16_t* bcol = Bs + (wn * 64 + tt * 16 + l16) * 40 + half * 8;
      v16bf b;
#pragma unroll
      for (int j = 0; j < 8; j++) { b[j] = bcol[j]; b[j + 8] = bcol[16 + j]; }
      acc[tt] = __builtin_amdgcn_wmma_f32_16x16x32_bf16(false, a, false, b, (short)0, acc[tt],
                                                        false, false);
    }
  }

  // C/D layout: VGPR r -> M = half*8 + r, lane%16 -> N. Branch-free epilogue.
#pragma unroll
  for (int tt = 0; tt < 4; tt++) {
    long ncol = bn + wn * 64 + tt * 16 + l16;
    float bv = BIAS ? bias[ncol] : 0.f;
#pragma unroll
    for (int r = 0; r < 8; r++) {
      long mrow = bm + wm * 16 + half * 8 + r;
      float v = acc[tt][r] + bv;
      if (GELU) {  // tanh-approx GELU (jax.nn.gelu default)
        float u = v;
        v = 0.5f * u * (1.f + tanhf(0.7978845608028654f * (u + 0.044715f * u * u * u)));
      }
      if (RESID) v += resid[mrow * (long)N + ncol];
      if (O32) out32[mrow * (long)N + ncol] = v;
      if (O16) out16[mrow * (long)N + ncol] = (bf16_t)v;
    }
  }
}

// ---------------------------------------------------------------------------
// LSH hashing: buckets[b,h,s] = argmax([qk@rot, -qk@rot]) over 64 candidates
// ---------------------------------------------------------------------------
__global__ void k_hash(const float* __restrict__ qk, const float* __restrict__ rotl,
                       int* __restrict__ buckets) {
  int bh = blockIdx.x;              // b*H + h
  int b = bh / kH, h = bh % kH;
  __shared__ float rs[kDH * kNBh];  // 8 KB: this head's rotation matrix
  for (int i = threadIdx.x; i < kDH * kNBh; i += blockDim.x)
    rs[i] = rotl[(long)h * kDH * kNBh + i];
  __syncthreads();
  for (int s = threadIdx.x; s < kS; s += blockDim.x) {
    const float* q = qk + ((long)(b * kS + s)) * kD + h * kDH;
    float qr[kDH];
#pragma unroll
    for (int d = 0; d < kDH; d++) qr[d] = q[d];
    float rdot[kNBh];
    for (int r = 0; r < kNBh; r++) {
      float acc = 0.f;
#pragma unroll
      for (int d = 0; d < kDH; d++) acc += qr[d] * rs[d * kNBh + r];
      rdot[r] = acc;
    }
    int best = 0; float bv = rdot[0];
    for (int j = 1; j < 2 * kNBh; j++) {
      float v = (j < kNBh) ? rdot[j] : -rdot[j - kNBh];
      if (v > bv) { bv = v; best = j; }    // strict > keeps first index (argmax)
    }
    buckets[(long)bh * kS + s] = best;
  }
}

// ---------------------------------------------------------------------------
// Stable sort by (bucket, pos): bitonic sort of 2048 packed keys in LDS.
// key = bucket*2048 + pos (unique), order[i] = key & 2047.
// ---------------------------------------------------------------------------
__global__ void k_sort(const int* __restrict__ buckets, int* __restrict__ order) {
  __shared__ int keys[kS];
  int bh = blockIdx.x;
  const int* bk = buckets + (long)bh * kS;
  for (int i = threadIdx.x; i < kS; i += blockDim.x) keys[i] = bk[i] * kS + i;
  __syncthreads();
  for (int k = 2; k <= kS; k <<= 1) {
    for (int j = k >> 1; j > 0; j >>= 1) {
      for (int i = threadIdx.x; i < kS; i += blockDim.x) {
        int ixj = i ^ j;
        if (ixj > i) {
          bool up = ((i & k) == 0);
          int a = keys[i], c = keys[ixj];
          if ((up && a > c) || (!up && a < c)) { keys[i] = c; keys[ixj] = a; }
        }
      }
      __syncthreads();
    }
  }
  for (int i = threadIdx.x; i < kS; i += blockDim.x) order[(long)bh * kS + i] = keys[i] & (kS - 1);
}

// ---------------------------------------------------------------------------
// Chunked LSH attention. One block per (b,h,chunk). Keys = prev chunk (wrap)
// + current chunk, l2-normalized (shared-QK). Output scattered directly to
// original positions in merged [B,S,H*DH] layout (fuses unsort + head-merge).
// ---------------------------------------------------------------------------
__launch_bounds__(128)
__global__ void k_lsh_attn(const float* __restrict__ qk, const float* __restrict__ vv,
                           const int* __restrict__ buckets, const int* __restrict__ order,
                           const int* __restrict__ amask, float* __restrict__ outm) {
  int blk = blockIdx.x;
  int n = blk % kNC; int bh = blk / kNC;
  int b = bh / kH, h = bh % kH;
  __shared__ bf16_t ks[2 * kC][kDH];   // normalized keys (16 KB)
  __shared__ float  vs[2 * kC][kDH];   // values (32 KB)
  __shared__ int be[2 * kC], ie[2 * kC], me[2 * kC];
  const int* ord = order + (long)bh * kS;
  int tid = threadIdx.x;

  if (tid < 2 * kC) {                   // one thread per key row
    int kk = tid;
    int si = (kk < kC) ? (((n - 1 + kNC) % kNC) * kC + kk) : (n * kC + (kk - kC));
    int pos = ord[si];
    ie[kk] = pos;
    be[kk] = buckets[(long)bh * kS + pos];
    me[kk] = amask[b * kS + pos];
    const float* kr = qk + ((long)(b * kS + pos)) * kD + h * kDH;
    const float* vr = vv + ((long)(b * kS + pos)) * kD + h * kDH;
    float tmp[kDH]; float nrm = 0.f;
#pragma unroll
    for (int d = 0; d < kDH; d++) { tmp[d] = kr[d]; nrm += tmp[d] * tmp[d]; }
    float inv = 1.f / (sqrtf(nrm) + 1e-6f);
#pragma unroll
    for (int d = 0; d < kDH; d++) { ks[kk][d] = (bf16_t)(tmp[d] * inv); vs[kk][d] = vr[d]; }
  }
  __syncthreads();

  if (tid < kC) {                       // one thread per query row
    int si = n * kC + tid;
    int qpos = ord[si];
    int bq = buckets[(long)bh * kS + qpos];
    const float* qr = qk + ((long)(b * kS + qpos)) * kD + h * kDH;
    float qreg[kDH];
#pragma unroll
    for (int d = 0; d < kDH; d++) qreg[d] = qr[d];
    float dots[2 * kC];
    float mx = -3.0e38f;
    for (int kk = 0; kk < 2 * kC; kk++) {
      float acc = 0.f;
#pragma unroll
      for (int d = 0; d < kDH; d++) acc += qreg[d] * (float)ks[kk][d];
      acc *= 0.125f;                              // 1/sqrt(64)
      if (bq != be[kk]) acc = -1e9f;              // cross-bucket
      if (me[kk] <= 0)  acc = -1e9f;              // attention mask
      if (qpos == ie[kk]) acc = -1e5f;            // soft self-mask (overrides)
      dots[kk] = acc;
      mx = fmaxf(mx, acc);
    }
    float sum = 0.f;
    for (int kk = 0; kk < 2 * kC; kk++) { float e = expf(dots[kk] - mx); dots[kk] = e; sum += e; }
    float invs = 1.f / sum;
    float* orow = outm + ((long)(b * kS + qpos)) * kD + h * kDH;
#pragma unroll
    for (int d = 0; d < kDH; d++) {
      float acc = 0.f;
      for (int kk = 0; kk < 2 * kC; kk++) acc += dots[kk] * vs[kk][d];
      orow[d] = acc * invs;
    }
  }
}

// ---------------------------------------------------------------------------
// Host: full pipeline on `stream`; scratch bump-allocated from d_ws.
// ---------------------------------------------------------------------------
extern "C" void kernel_launch(void* const* d_in, const int* in_sizes, int n_in,
                              void* d_out, int out_size, void* d_ws, size_t ws_size,
                              hipStream_t stream) {
  (void)in_sizes; (void)n_in; (void)out_size; (void)ws_size;
  const int*   ids   = (const int*)d_in[0];
  const int*   amask = (const int*)d_in[1];
  const float* emb   = (const float*)d_in[2];
  const float* pose  = (const float*)d_in[3];
  const float* rot   = (const float*)d_in[4];
  const float* Wqk   = (const float*)d_in[5];
  const float* Wv    = (const float*)d_in[6];
  const float* Wo    = (const float*)d_in[7];
  const float* ln1g  = (const float*)d_in[8];
  const float* ln1b  = (const float*)d_in[9];
  const float* ln2g  = (const float*)d_in[10];
  const float* ln2b  = (const float*)d_in[11];
  const float* W1    = (const float*)d_in[12];
  const float* b1    = (const float*)d_in[13];
  const float* W2    = (const float*)d_in[14];
  const float* b2    = (const float*)d_in[15];
  const float* lnfg  = (const float*)d_in[16];
  const float* lnfb  = (const float*)d_in[17];
  const float* Wlm   = (const float*)d_in[18];
  const float* blm   = (const float*)d_in[19];
  float* out = (float*)d_out;

  char* ws = (char*)d_ws;
  size_t off = 0;
  auto alloc = [&](size_t bytes) -> void* {
    void* p = ws + off;
    off = (off + bytes + 255) & ~(size_t)255;
    return p;
  };

  const long nWqk = (long)kL * kD * kD;        // 2*768*768
  const long nW1  = (long)kL * kD * kF;        // 2*768*3072
  const long nWlm = (long)kD * kV;             // 768*16384
  const long nMD  = (long)kM * kD;             // 4096*768
  const long nMF  = (long)kM * kF;             // 4096*3072
  const long nBHS = (long)kB * kH * kS;

  bf16_t* wqk16 = (bf16_t*)alloc(nWqk * 2);
  bf16_t* wv16  = (bf16_t*)alloc(nWqk * 2);
  bf16_t* wo16  = (bf16_t*)alloc(nWqk * 2);
  bf16_t* w116  = (bf16_t*)alloc(nW1 * 2);
  bf16_t* w216  = (bf16_t*)alloc(nW1 * 2);
  bf16_t* wlm16 = (bf16_t*)alloc(nWlm * 2);
  float*  x     = (float*)alloc(nMD * 4);
  float*  x2    = (float*)alloc(nMD * 4);
  bf16_t* h16   = (bf16_t*)alloc(nMD * 2);
  float*  qkb   = (float*)alloc(nMD * 4);
  float*  vb    = (float*)alloc(nMD * 4);
  float*  attn  = (float*)alloc(nMD * 4);
  bf16_t* attn16= (bf16_t*)alloc(nMD * 2);
  bf16_t* ffn16 = (bf16_t*)alloc(nMF * 2);
  int*    bkts  = (int*)alloc(nBHS * 4);
  int*    order = (int*)alloc(nBHS * 4);

  auto conv = [&](const float* src, bf16_t* dst, long n) {
    int blocks = (int)((n + 256L * 8 - 1) / (256L * 8));
    k_f32_to_bf16<<<blocks, 256, 0, stream>>>(src, dst, n);
  };
  // templated dispatch: (BIAS, GELU, RESID, O32, O16)
  auto gemm = [&](const bf16_t* Aa, const bf16_t* Ww, const float* bias, const float* resid,
                  float* o32, bf16_t* o16, int M_, int N_, int K_, bool gelu) {
    dim3 g(M_ / 64, N_ / 128);
    if (gelu)
      k_gemm_bf16<true, true, false, false, true><<<g, 256, 0, stream>>>(Aa, Ww, bias, resid, o32, o16, M_, N_, K_);
    else if (bias && resid)
      k_gemm_bf16<true, false, true, true, false><<<g, 256, 0, stream>>>(Aa, Ww, bias, resid, o32, o16, M_, N_, K_);
    else if (bias)
      k_gemm_bf16<true, false, false, true, false><<<g, 256, 0, stream>>>(Aa, Ww, bias, resid, o32, o16, M_, N_, K_);
    else if (resid)
      k_gemm_bf16<false, false, true, true, false><<<g, 256, 0, stream>>>(Aa, Ww, bias, resid, o32, o16, M_, N_, K_);
    else
      k_gemm_bf16<false, false, false, true, false><<<g, 256, 0, stream>>>(Aa, Ww, bias, resid, o32, o16, M_, N_, K_);
  };

  // weight conversions (once per launch; deterministic)
  conv(Wqk, wqk16, nWqk);
  conv(Wv,  wv16,  nWqk);
  conv(Wo,  wo16,  nWqk);
  conv(W1,  w116,  nW1);
  conv(W2,  w216,  nW1);
  conv(Wlm, wlm16, nWlm);

  // x = emb[ids] + pos
  k_embed<<<kM, 256, 0, stream>>>(ids, emb, pose, x);

  float* xc = x;
  float* xn = x2;
  for (int l = 0; l < kL; l++) {
    // --- attention sub-block ---
    k_layernorm<<<kM, 256, 0, stream>>>(xc, ln1g + (long)l * kD, ln1b + (long)l * kD, nullptr, h16);
    gemm(h16, wqk16 + (long)l * kD * kD, nullptr, nullptr, qkb, nullptr, kM, kD, kD, false);
    gemm(h16, wv16  + (long)l * kD * kD, nullptr, nullptr, vb,  nullptr, kM, kD, kD, false);
    k_hash<<<kB * kH, 256, 0, stream>>>(qkb, rot + (long)l * kH * kDH * kNBh, bkts);
    k_sort<<<kB * kH, 256, 0, stream>>>(bkts, order);
    k_lsh_attn<<<kB * kH * kNC, 128, 0, stream>>>(qkb, vb, bkts, order, amask, attn);
    conv(attn, attn16, nMD);
    gemm(attn16, wo16 + (long)l * kD * kD, nullptr, /*resid=*/xc, xn, nullptr, kM, kD, kD, false);
    { float* t = xc; xc = xn; xn = t; }
    // --- FFN sub-block ---
    k_layernorm<<<kM, 256, 0, stream>>>(xc, ln2g + (long)l * kD, ln2b + (long)l * kD, nullptr, h16);
    gemm(h16, w116 + (long)l * kD * kF, b1 + (long)l * kF, nullptr, nullptr, ffn16, kM, kF, kD, true);
    gemm(ffn16, w216 + (long)l * kF * kD, b2 + (long)l * kD, /*resid=*/xc, xn, nullptr, kM, kD, kF, false);
    { float* t = xc; xc = xn; xn = t; }
  }

  // final LN + LM head
  k_layernorm<<<kM, 256, 0, stream>>>(xc, lnfg, lnfb, nullptr, h16);
  gemm(h16, wlm16, blm, nullptr, out, nullptr, kM, kV, kD, false);
}
